// ActionDecoder_2714419331180
// MI455X (gfx1250) — matrix operand order repaired and checked by reading, not compiled, not run on hardware
//
#include <hip/hip_runtime.h>
#include <hip/hip_bf16.h>

typedef _Float16 v16h __attribute__((ext_vector_type(16)));
typedef _Float16 v8h  __attribute__((ext_vector_type(8)));
typedef float    v8f  __attribute__((ext_vector_type(8)));
typedef __attribute__((address_space(3))) _Float16 lds_f16_t;

#define B_   64
#define T_   512
#define V_   512
#define E_   128
#define IMG_ 1024
#define H_   512
#define A_   512
#define G4_  2048   // 4*H
#define IN_  1152   // IMG+E
#define NWG  16

// ---------- WMMA helpers (CDNA5 gfx1250, wave32) ----------

__device__ __forceinline__ v8f wmma_f16(v16h a, v16h b, v8f c) {
  // D = A(16x32 f16) * B(32x16 f16) + C(16x16 f32)
  return __builtin_amdgcn_wmma_f32_16x16x32_f16(false, a, false, b, (short)0, c, false, false);
}

// A-fragment (16x32 f16) from row-major f16 matrix [rows][ld] (global or LDS).
// ISA 7.12.2: lane L: M = L%16; K-base = (L/16)*8; v0..3 = K base..base+7, v4..7 = K base+16..base+23.
__device__ __forceinline__ v16h load_a_h(const _Float16* base, int row0, int ld, int k0, int lane) {
  int r  = row0 + (lane & 15);
  int kb = k0 + ((lane >> 4) << 3);
  const _Float16* p = base + (size_t)r * ld + kb;
  v8h lo = *(const v8h*)(p);
  v8h hi = *(const v8h*)(p + 16);
  v16h a;
  #pragma unroll
  for (int i = 0; i < 8; ++i) { a[i] = lo[i]; a[8 + i] = hi[i]; }
  return a;
}

// Same A layout, converting from f32 source on the fly.
__device__ __forceinline__ v16h load_a_f(const float* base, int row0, int ld, int k0, int lane) {
  int r  = row0 + (lane & 15);
  int kb = k0 + ((lane >> 4) << 3);
  const float* p = base + (size_t)r * ld + kb;
  v16h a;
  #pragma unroll
  for (int i = 0; i < 8; ++i) { a[i] = (_Float16)p[i]; a[8 + i] = (_Float16)p[16 + i]; }
  return a;
}

// B-fragment (32x16 f16) for computing X @ W^T with W row-major [N][K].
// Same K order as A: lane L: N = L%16; K-base = (L/16)*8; v0..3 = K base..base+7, v4..7 = K base+16..base+23.
__device__ __forceinline__ v16h load_b_f(const float* base, int n0, int ld, int k0, int lane) {
  int r  = n0 + (lane & 15);
  int kb = k0 + ((lane >> 4) << 3);
  const float* p = base + (size_t)r * ld + kb;
  v16h b;
  #pragma unroll
  for (int i = 0; i < 8; ++i) { b[i] = (_Float16)p[i]; b[8 + i] = (_Float16)p[16 + i]; }
  return b;
}

// ---------- grid barrier (sense-reversing, 16 co-resident WGs) ----------

__device__ __forceinline__ void grid_barrier(unsigned* counter, unsigned* phase, int s, unsigned nblocks) {
  __syncthreads();
  if (threadIdx.x == 0) {
    __threadfence();  // release: make h/hn stores visible device-wide
    if (atomicAdd(counter, 1u) == nblocks - 1u) {
      atomicExch(counter, 0u);
      __threadfence();
      __hip_atomic_store(phase, (unsigned)s, __ATOMIC_RELEASE, __HIP_MEMORY_SCOPE_AGENT);
    } else {
      while (__hip_atomic_load(phase, __ATOMIC_ACQUIRE, __HIP_MEMORY_SCOPE_AGENT) != (unsigned)s) {
        __builtin_amdgcn_s_sleep(2);
      }
    }
    __threadfence();  // acquire
  }
  __syncthreads();
}

__global__ void k_init(unsigned* bar) { bar[0] = 0u; bar[1] = 0u; }

// ---------- kernel 1: emb_proj[V][4H] = emb @ W_ih[:,IMG:]^T ----------

__global__ __launch_bounds__(256) void k_emb_proj(const float* __restrict__ emb,
                                                  const float* __restrict__ W_ih,
                                                  float* __restrict__ emb_proj) {
  int gid  = blockIdx.x * 8 + (threadIdx.x >> 5);  // 4096 waves: 32 M-tiles x 128 N-tiles
  int lane = threadIdx.x & 31;
  int m0 = (gid & 31) * 16;
  int n0 = (gid >> 5) * 16;
  v8f acc = {};
  #pragma unroll
  for (int ks = 0; ks < 4; ++ks) {                  // K = E = 128
    v16h a = load_a_f(emb, m0, E_, ks * 32, lane);
    v16h b = load_b_f(W_ih + IMG_, n0, IN_, ks * 32, lane);
    acc = wmma_f16(a, b, acc);
  }
  int rb  = m0 + ((lane >> 4) << 3);
  int col = n0 + (lane & 15);
  #pragma unroll
  for (int v = 0; v < 8; ++v) emb_proj[(size_t)(rb + v) * G4_ + col] = acc[v];
}

// ---------- kernel 2: img_proj[B][4H] = x1 @ W_ih[:,:IMG]^T + b_ih + b_hh ----------

__global__ __launch_bounds__(256) void k_img_proj(const float* __restrict__ x1,
                                                  const float* __restrict__ W_ih,
                                                  const float* __restrict__ b_ih,
                                                  const float* __restrict__ b_hh,
                                                  float* __restrict__ img_proj) {
  int gid  = blockIdx.x * 8 + (threadIdx.x >> 5);  // 512 waves: 4 M-tiles x 128 N-tiles
  int lane = threadIdx.x & 31;
  int m0 = (gid & 3) * 16;
  int n0 = (gid >> 2) * 16;
  v8f acc = {};
  for (int ks = 0; ks < 32; ++ks) {                 // K = IMG = 1024
    v16h a = load_a_f(x1, m0, IMG_, ks * 32, lane);
    v16h b = load_b_f(W_ih, n0, IN_, ks * 32, lane);
    acc = wmma_f16(a, b, acc);
  }
  int rb  = m0 + ((lane >> 4) << 3);
  int col = n0 + (lane & 15);
  float bias = b_ih[col] + b_hh[col];
  #pragma unroll
  for (int v = 0; v < 8; ++v) img_proj[(size_t)(rb + v) * G4_ + col] = acc[v] + bias;
}

// ---------- kernel 3: persistent LSTM over T=512 steps ----------
// 16 WGs; WG wg owns hidden units [32*wg, 32*wg+32): a 128-column gate slice of W_hh
// whose WMMA B-fragments live in registers for the whole kernel. h_prev is staged
// into LDS each step via CDNA5 async global->LDS DMA (ASYNCcnt), then consumed by
// WMMA A-fragment ds loads.

__global__ __launch_bounds__(256) void k_lstm(const int* __restrict__ x2,
                                              const int* __restrict__ lens,
                                              const float* __restrict__ W_hh,
                                              const float* __restrict__ img_proj,
                                              const float* __restrict__ emb_proj,
                                              _Float16* __restrict__ h_buf,   // [2][B][H] ping-pong
                                              _Float16* __restrict__ hn,      // [B][H]
                                              unsigned* __restrict__ bar) {
  __shared__ float gates[B_][128];     // [batch][i(32) f(32) g(32) o(32)] raw h@W_hh^T
  __shared__ float cbuf[B_][32];       // cell state for this WG's 32 units
  extern __shared__ _Float16 h_lds[];  // [B_][H_] = 64KB dynamic: staged h_prev
  const int wg   = blockIdx.x;         // 0..15
  const int tid  = threadIdx.x;
  const int wave = tid >> 5;           // 0..7: wave w -> local gate cols [16w,16w+16)
  const int lane = tid & 31;

  // Preload step-invariant W_hh fragments: wave's 16 gate rows, all K=512 -> 16 v16h (128 VGPRs).
  const int gate = wave >> 1;                                 // 0:i 1:f 2:g 3:o
  const int wrow = gate * H_ + wg * 32 + ((wave & 1) << 4);   // W_hh row base
  v16h bfrag[16];
  #pragma unroll
  for (int ks = 0; ks < 16; ++ks)
    bfrag[ks] = load_b_f(W_hh, wrow, H_, ks * 32, lane);

  // zero c state and our slice of h0
  for (int i = tid; i < B_ * 32; i += 256) {
    int b = i >> 5, j = i & 31;
    cbuf[b][j] = 0.0f;
    h_buf[(size_t)b * H_ + wg * 32 + j] = (_Float16)0.0f;
  }
  int bs = 0;
  bs ^= 1; grid_barrier(bar, bar + 1, bs, NWG);   // h0 visible everywhere

  for (int t = 0; t < T_; ++t) {
    const _Float16* h_prev = h_buf + (size_t)(t & 1) * (B_ * H_);
    _Float16*       h_next = h_buf + (size_t)((t + 1) & 1) * (B_ * H_);

    // --- stage h_prev (64KB) into LDS: 4096 x 16B async DMA transfers ---
    #pragma unroll
    for (int c = 0; c < 16; ++c) {
      int e = (tid + c * 256) * 8;                       // half-element index, 16B aligned
      unsigned ldst = (unsigned)(size_t)(lds_f16_t*)(h_lds + e);
      const _Float16* gsrc = h_prev + e;
      asm volatile("global_load_async_to_lds_b128 %0, %1, off"
                   :: "v"(ldst), "v"(gsrc) : "memory");
    }
    asm volatile("s_wait_asynccnt 0x0" ::: "memory");
    __syncthreads();

    // gates slice = h_lds(64x512) @ W_hh_slice^T(512x128): 4 M-tiles per wave, K=512
    v8f acc[4] = {};
    #pragma unroll
    for (int ks = 0; ks < 16; ++ks) {
      #pragma unroll
      for (int m = 0; m < 4; ++m) {
        v16h a = load_a_h(h_lds, m * 16, H_, ks * 32, lane);
        acc[m] = wmma_f16(a, bfrag[ks], acc[m]);
      }
    }
    #pragma unroll
    for (int m = 0; m < 4; ++m) {
      int rb  = m * 16 + ((lane >> 4) << 3);
      int col = wave * 16 + (lane & 15);
      #pragma unroll
      for (int v = 0; v < 8; ++v) gates[rb + v][col] = acc[m][v];
    }
    __syncthreads();

    // pointwise LSTM update for this WG's 32 units, all 64 batches (f32 math)
    for (int i = tid; i < B_ * 32; i += 256) {
      int b = i >> 5, j = i & 31;
      int unit = wg * 32 + j;
      int tok  = x2[b * T_ + t];
      const float* ip = img_proj + (size_t)b   * G4_;
      const float* ep = emb_proj + (size_t)tok * G4_;
      float gi = gates[b][j]      + ip[unit]           + ep[unit];
      float gf = gates[b][32 + j] + ip[H_ + unit]      + ep[H_ + unit];
      float gg = gates[b][64 + j] + ip[2 * H_ + unit]  + ep[2 * H_ + unit];
      float go = gates[b][96 + j] + ip[3 * H_ + unit]  + ep[3 * H_ + unit];
      float ii = 1.0f / (1.0f + __expf(-gi));
      float ff = 1.0f / (1.0f + __expf(-gf));
      float g2 = tanhf(gg);
      float oo = 1.0f / (1.0f + __expf(-go));
      float c  = ff * cbuf[b][j] + ii * g2;
      cbuf[b][j] = c;
      float h = oo * tanhf(c);
      h_next[(size_t)b * H_ + unit] = (_Float16)h;
      if (t == lens[b] - 1) hn[(size_t)b * H_ + unit] = (_Float16)h;  // packed-seq hn
    }
    bs ^= 1; grid_barrier(bar, bar + 1, bs, NWG);
  }
}

// ---------- kernel 4: nt = hn @ W_act^T + b_act ----------

__global__ __launch_bounds__(256) void k_action(const _Float16* __restrict__ hn,
                                                const float* __restrict__ W_act,
                                                const float* __restrict__ b_act,
                                                float* __restrict__ out) {
  int gid  = blockIdx.x * 8 + (threadIdx.x >> 5);  // 128 waves: 4 M-tiles x 32 N-tiles
  int lane = threadIdx.x & 31;
  int m0 = (gid & 3) * 16;
  int n0 = (gid >> 2) * 16;
  v8f acc = {};
  #pragma unroll
  for (int ks = 0; ks < 16; ++ks) {                 // K = H = 512
    v16h a = load_a_h(hn, m0, H_, ks * 32, lane);
    v16h b = load_b_f(W_act, n0, H_, ks * 32, lane);
    acc = wmma_f16(a, b, acc);
  }
  int rb  = m0 + ((lane >> 4) << 3);
  int col = n0 + (lane & 15);
  float bias = b_act[col];
  #pragma unroll
  for (int v = 0; v < 8; ++v) out[(size_t)(rb + v) * A_ + col] = acc[v] + bias;
}

// ---------- host ----------

extern "C" void kernel_launch(void* const* d_in, const int* in_sizes, int n_in,
                              void* d_out, int out_size, void* d_ws, size_t ws_size,
                              hipStream_t stream) {
  const float* x1    = (const float*)d_in[0];
  const int*   x2    = (const int*)  d_in[1];
  const int*   lens  = (const int*)  d_in[2];
  const float* emb   = (const float*)d_in[3];
  const float* W_ih  = (const float*)d_in[4];
  const float* W_hh  = (const float*)d_in[5];
  const float* b_ih  = (const float*)d_in[6];
  const float* b_hh  = (const float*)d_in[7];
  const float* W_act = (const float*)d_in[8];
  const float* b_act = (const float*)d_in[9];
  float* out = (float*)d_out;

  // workspace layout (~4.92 MB)
  float*    emb_proj = (float*)d_ws;                            // V x 4H f32
  float*    img_proj = emb_proj + (size_t)V_ * G4_;             // B x 4H f32
  _Float16* h_buf    = (_Float16*)(img_proj + (size_t)B_ * G4_);// 2 x B x H f16
  _Float16* hn       = h_buf + 2 * B_ * H_;                     // B x H f16
  unsigned* bar      = (unsigned*)(hn + B_ * H_);               // 2 words

  hipLaunchKernelGGL(k_init,     dim3(1),   dim3(1),   0, stream, bar);
  hipLaunchKernelGGL(k_emb_proj, dim3(512), dim3(256), 0, stream, emb, W_ih, emb_proj);
  hipLaunchKernelGGL(k_img_proj, dim3(64),  dim3(256), 0, stream, x1, W_ih, b_ih, b_hh, img_proj);
  hipLaunchKernelGGL(k_lstm,     dim3(NWG), dim3(256), B_ * H_ * sizeof(_Float16), stream,
                     x2, lens, W_hh, img_proj, emb_proj, h_buf, hn, bar);
  hipLaunchKernelGGL(k_action,   dim3(16),  dim3(256), 0, stream, hn, W_act, b_act, out);
}
